// GraphSAGENet_80556406604176
// MI455X (gfx1250) — matrix operand order repaired and checked by reading, not compiled, not run on hardware
//
#include <hip/hip_runtime.h>

#define N_NODES 50000
#define N_EDGES 800000
#define IN_FEATS 96
#define HIDDEN 48

typedef __attribute__((ext_vector_type(16))) __bf16 v16bf;
typedef __attribute__((ext_vector_type(8)))  __bf16 v8bf;
typedef __attribute__((ext_vector_type(8)))  float  v8f;

static __device__ __forceinline__ __bf16 to_bf16(float f) { return (__bf16)f; }

// ---------------- utility kernels ----------------

__global__ void zero_f32_kernel(float* __restrict__ p, int n) {
    int i = blockIdx.x * blockDim.x + threadIdx.x;
    if (i < n) p[i] = 0.0f;
}

__global__ void deg_kernel(const int* __restrict__ dst, float* __restrict__ deg) {
    int e = blockIdx.x * blockDim.x + threadIdx.x;
    if (e < N_EDGES) atomicAdd(&deg[dst[e]], 1.0f);
}

__global__ void deg_inv_kernel(float* __restrict__ deg) {
    int i = blockIdx.x * blockDim.x + threadIdx.x;
    if (i < N_NODES) deg[i] = 1.0f / fmaxf(deg[i], 1.0f);
}

// one thread per (edge, feature): gather x[src], scatter-add into agg[dst]
__global__ void scatter_kernel(const float* __restrict__ x,
                               const int* __restrict__ src,
                               const int* __restrict__ dst,
                               float* __restrict__ agg, int F) {
    long long t = (long long)blockIdx.x * blockDim.x + threadIdx.x;
    long long total = (long long)N_EDGES * F;
    if (t >= total) return;
    int e = (int)(t / F);
    int f = (int)(t % F);
    float v = x[(long long)src[e] * F + f];
    atomicAdd(&agg[(long long)dst[e] * F + f], v);
}

// fp32 -> bf16 conversion of self features + mean-scaled aggregate, K-padded
__global__ void convert_kernel(const float* __restrict__ x,
                               const float* __restrict__ agg,
                               const float* __restrict__ deg_inv,
                               __bf16* __restrict__ xb, __bf16* __restrict__ ab,
                               int F, int Fp) {
    int t = blockIdx.x * blockDim.x + threadIdx.x;
    int total = N_NODES * Fp;
    if (t >= total) return;
    int n = t / Fp;
    int f = t % Fp;
    float xv = 0.0f, av = 0.0f;
    if (f < F) {
        xv = x[(long long)n * F + f];
        av = agg[(long long)n * F + f] * deg_inv[n];
    }
    xb[t] = to_bf16(xv);
    ab[t] = to_bf16(av);
}

// Pack weight [F x HIDDEN] (row-major fp32) into WMMA B-fragment lane order, bf16,
// zero-padded to KT*32 rows. Layout: Wp[((nt*KT + kt)*32 + lane)*16 + e]
__global__ void pack_w_kernel(const float* __restrict__ W, __bf16* __restrict__ Wp,
                              int F, int KT) {
    int t = blockIdx.x * blockDim.x + threadIdx.x;
    int total = (HIDDEN / 16) * KT * 32 * 16;
    if (t >= total) return;
    int e    = t & 15;
    int lane = (t >> 4) & 31;
    int rest = t >> 9;          // nt*KT + kt
    int kt   = rest % KT;
    int nt   = rest / KT;
    int h    = lane >> 4;
    int l16  = lane & 15;
    // B-matrix 32x16 bf16 fragment: element e holds K = h*8 + e (e<8), 16 + h*8 + (e-8) (e>=8), N = l16
    int K = kt * 32 + ((e < 8) ? (h * 8 + e) : (16 + h * 8 + (e - 8)));
    int N = nt * 16 + l16;
    float v = (K < F) ? W[(long long)K * HIDDEN + N] : 0.0f;
    Wp[t] = to_bf16(v);
}

// Fused SAGE layer: out = relu(Xb @ Wself + Ab @ Wneigh + bias)
// One wave computes one 16x16 output tile via v_wmma_f32_16x16x32_bf16.
__global__ __launch_bounds__(256)
void sage_wmma_kernel(const __bf16* __restrict__ xb, const __bf16* __restrict__ ab,
                      const __bf16* __restrict__ wpS, const __bf16* __restrict__ wpN,
                      const float* __restrict__ bias, float* __restrict__ out,
                      int KT, int Fp) {
    const int NT = HIDDEN / 16;      // 3
    const int MT = N_NODES / 16;     // 3125 (exact)
    int wave = (int)((blockIdx.x * blockDim.x + threadIdx.x) >> 5);
    int lane = threadIdx.x & 31;
    if (wave >= MT * NT) return;     // wave-uniform: EXEC all-ones inside
    int mt = wave / NT;
    int nt = wave % NT;
    int h   = lane >> 4;
    int l16 = lane & 15;

    // A fragment: lane (h,l16) holds row M=l16; elements 0..7 -> K = h*8+0..7,
    // elements 8..15 -> K = 16 + h*8 + 0..7 (two contiguous 16B runs).
    const __bf16* xrow = xb + (long long)(mt * 16 + l16) * Fp;
    const __bf16* arow = ab + (long long)(mt * 16 + l16) * Fp;

    v8f acc = {};
    for (int kt = 0; kt < KT; ++kt) {
        int kk = kt * 32;
        union { v16bf v; v8bf p[2]; } aS, aN, bS, bN;
        aS.p[0] = *(const v8bf*)(xrow + kk + h * 8);
        aS.p[1] = *(const v8bf*)(xrow + kk + 16 + h * 8);
        aN.p[0] = *(const v8bf*)(arow + kk + h * 8);
        aN.p[1] = *(const v8bf*)(arow + kk + 16 + h * 8);
        const __bf16* ws = wpS + (((nt * KT + kt) * 32 + lane) << 4);
        const __bf16* wn = wpN + (((nt * KT + kt) * 32 + lane) << 4);
        bS.p[0] = *(const v8bf*)(ws);
        bS.p[1] = *(const v8bf*)(ws + 8);
        bN.p[0] = *(const v8bf*)(wn);
        bN.p[1] = *(const v8bf*)(wn + 8);
        acc = __builtin_amdgcn_wmma_f32_16x16x32_bf16(
            false, aS.v, false, bS.v, (short)0, acc, false, false);
        acc = __builtin_amdgcn_wmma_f32_16x16x32_bf16(
            false, aN.v, false, bN.v, (short)0, acc, false, false);
    }

    // C/D layout: VGPR r -> M = h*8 + r, N = l16
    int ncol = nt * 16 + l16;
    float bv = bias[ncol];
#pragma unroll
    for (int r = 0; r < 8; ++r) {
        int m = mt * 16 + h * 8 + r;
        float v = acc[r] + bv;
        out[(long long)m * HIDDEN + ncol] = fmaxf(v, 0.0f);
    }
}

// prediction head: out[n] = h[n,:] . w_pred + b_pred
__global__ void head_kernel(const float* __restrict__ h,
                            const float* __restrict__ wp,
                            const float* __restrict__ bp,
                            float* __restrict__ out) {
    int n = blockIdx.x * blockDim.x + threadIdx.x;
    if (n >= N_NODES) return;
    float s = bp[0];
#pragma unroll
    for (int f = 0; f < HIDDEN; ++f) s += h[(long long)n * HIDDEN + f] * wp[f];
    out[n] = s;
}

// ---------------- host driver ----------------

static inline int cdiv(long long a, long long b) { return (int)((a + b - 1) / b); }

extern "C" void kernel_launch(void* const* d_in, const int* in_sizes, int n_in,
                              void* d_out, int out_size, void* d_ws, size_t ws_size,
                              hipStream_t stream) {
    (void)in_sizes; (void)n_in; (void)out_size; (void)ws_size;
    const float* x        = (const float*)d_in[0];
    const int*   eidx     = (const int*)d_in[1];
    const int*   src      = eidx;
    const int*   dst      = eidx + N_EDGES;
    const float* w_self[3]  = { (const float*)d_in[2], (const float*)d_in[5], (const float*)d_in[8] };
    const float* w_neigh[3] = { (const float*)d_in[3], (const float*)d_in[6], (const float*)d_in[9] };
    const float* bias[3]    = { (const float*)d_in[4], (const float*)d_in[7], (const float*)d_in[10] };
    const float* w_pred  = (const float*)d_in[11];
    const float* b_pred  = (const float*)d_in[12];
    float* out = (float*)d_out;

    // workspace carve-up (256B aligned)
    char* ws = (char*)d_ws;
    size_t off = 0;
    auto carve = [&](size_t bytes) -> char* {
        char* p = ws + off;
        off = (off + bytes + 255) & ~(size_t)255;
        return p;
    };
    float*  deg  = (float*)carve((size_t)N_NODES * 4);
    float*  agg  = (float*)carve((size_t)N_NODES * IN_FEATS * 4);
    __bf16* xb   = (__bf16*)carve((size_t)N_NODES * IN_FEATS * 2);
    __bf16* ab   = (__bf16*)carve((size_t)N_NODES * IN_FEATS * 2);
    float*  hA   = (float*)carve((size_t)N_NODES * HIDDEN * 4);
    float*  hB   = (float*)carve((size_t)N_NODES * HIDDEN * 4);
    __bf16* wpS  = (__bf16*)carve((size_t)3 * 3 * 32 * 16 * 2);
    __bf16* wpN  = (__bf16*)carve((size_t)3 * 3 * 32 * 16 * 2);

    const int B = 256;

    // degrees -> deg_inv (once)
    zero_f32_kernel<<<cdiv(N_NODES, B), B, 0, stream>>>(deg, N_NODES);
    deg_kernel<<<cdiv(N_EDGES, B), B, 0, stream>>>(dst, deg);
    deg_inv_kernel<<<cdiv(N_NODES, B), B, 0, stream>>>(deg);

    const float* layer_in = x;
    float* layer_out[3] = { hA, hB, hA };
    for (int L = 0; L < 3; ++L) {
        int F  = (L == 0) ? IN_FEATS : HIDDEN;   // 96 / 48 / 48
        int Fp = (L == 0) ? IN_FEATS : 64;       // K padded to multiple of 32
        int KT = Fp / 32;                        // 3 / 2 / 2

        // agg = segment_sum(x[src]) at dst
        zero_f32_kernel<<<cdiv((long long)N_NODES * F, B), B, 0, stream>>>(agg, N_NODES * F);
        scatter_kernel<<<cdiv((long long)N_EDGES * F, B), B, 0, stream>>>(layer_in, src, dst, agg, F);

        // bf16 conversion (self + mean aggregate), K-padded with zeros
        convert_kernel<<<cdiv((long long)N_NODES * Fp, B), B, 0, stream>>>(
            layer_in, agg, deg, xb, ab, F, Fp);

        // pack both weight matrices into WMMA B-fragment order
        int wtot = (HIDDEN / 16) * KT * 32 * 16;
        pack_w_kernel<<<cdiv(wtot, B), B, 0, stream>>>(w_self[L], wpS, F, KT);
        pack_w_kernel<<<cdiv(wtot, B), B, 0, stream>>>(w_neigh[L], wpN, F, KT);

        // fused dual-GEMM + bias + relu via v_wmma_f32_16x16x32_bf16
        int nwaves = (N_NODES / 16) * (HIDDEN / 16);   // 9375
        sage_wmma_kernel<<<cdiv((long long)nwaves * 32, B), B, 0, stream>>>(
            xb, ab, wpS, wpN, bias[L], layer_out[L], KT, Fp);

        layer_in = layer_out[L];
    }

    head_kernel<<<cdiv(N_NODES, B), B, 0, stream>>>(hA, w_pred, b_pred, out);
}
